// NeuralPoissonMultiResMLP_15968688406826
// MI455X (gfx1250) — matrix operand order repaired and checked by reading, not compiled, not run on hardware
//
#include <hip/hip_runtime.h>
#include <hip/hip_bf16.h>
#include <math.h>

typedef __attribute__((ext_vector_type(16))) _Float16 v16h;
typedef __attribute__((ext_vector_type(8)))  float    v8f;

namespace cfg {
constexpr int N      = 1 << 20;      // points
constexpr int WAVES  = 8;            // waves per block (wave32)
constexpr int PTS_W  = 16;           // points per wave (WMMA M)
constexpr int PTS_B  = WAVES * PTS_W;   // 128
constexpr int BLOCKS = N / PTS_B;       // 8192 (exact)
constexpr int TS     = 136;          // f16 row stride of act tiles (272B, 16B aligned)

// dynamic shared memory layout (bytes)
constexpr int SZ_W2F  = 8 * 4 * 32 * 16 * 2;        // fragment-ready W2  (32 KB)
constexpr int OFF_W2F = 0;
constexpr int OFF_W2T = OFF_W2F + SZ_W2F;           // fragment-ready W2^T (32 KB)
constexpr int OFF_W1  = OFF_W2T + SZ_W2F;           // 5x128 f32
constexpr int OFF_B1  = OFF_W1 + 5 * 128 * 4;
constexpr int OFF_B2  = OFF_B1 + 128 * 4;
constexpr int OFF_W3  = OFF_B2 + 128 * 4;
constexpr int OFF_WV  = OFF_W3 + 128 * 4;
// per-wave scratch
constexpr int WB_ACT  = 0;                          // 16 x TS f16 (h1 -> gz2 -> gz1)
constexpr int WB_C1   = WB_ACT + 16 * TS * 2;       // 16 x TS f16 (30*cos(30*z1))
constexpr int WB_XF   = WB_C1 + 16 * TS * 2;        // 16 x 8 f32 point features
constexpr int WB_RED  = WB_XF + 16 * 8 * 4;         // 32 x 8 f32 reductions
constexpr int WVB     = WB_RED + 32 * 8 * 4;        // 10240 B per wave
constexpr int SMEM    = OFF_WV + WAVES * WVB;       // ~148 KB total
}

union F16Frag { uint4 u[2]; v16h v; };

// A fragment: 16x32 f16, chunk base kb in {0,32,64,96}; per ISA layout:
// lane half kh: elems j=0..7 -> K=kb+kh*8+j, j=8..15 -> K=kb+16+kh*8+(j-8)
__device__ inline v16h load_a_frag(const _Float16* tile, int m, int kb, int kh) {
    const _Float16* base = tile + m * cfg::TS + kb + kh * 8;
    F16Frag f;
    f.u[0] = *(const uint4*)(base);
    f.u[1] = *(const uint4*)(base + 16);
    return f.v;
}

// B fragment: pre-staged contiguous 16 f16 (32B) per (ntile,kstep,lane)
__device__ inline v16h load_b_frag(const _Float16* buf, int nt, int ks, int lane) {
    const uint4* p = (const uint4*)(buf + ((((nt << 2) + ks) << 5) + lane) * 16);
    F16Frag f;
    f.u[0] = p[0];
    f.u[1] = p[1];
    return f.v;
}

__device__ inline v8f wmma16(v16h a, v16h b, v8f c) {
    return __builtin_amdgcn_wmma_f32_16x16x32_f16(false, a, false, b, (short)0, c,
                                                  false, false);
}

__global__ __launch_bounds__(256, 1)
void siren_sdf_fused(const float* __restrict__ pos, const float* __restrict__ grid,
                     const float* __restrict__ W1, const float* __restrict__ b1,
                     const float* __restrict__ W2, const float* __restrict__ b2,
                     const float* __restrict__ W3, const float* __restrict__ b3,
                     float* __restrict__ out) {
    extern __shared__ __align__(16) char smem[];
    _Float16* sW2f = (_Float16*)(smem + cfg::OFF_W2F);
    _Float16* sW2t = (_Float16*)(smem + cfg::OFF_W2T);
    float*    sW1  = (float*)(smem + cfg::OFF_W1);
    float*    sB1  = (float*)(smem + cfg::OFF_B1);
    float*    sB2  = (float*)(smem + cfg::OFF_B2);
    float*    sW3  = (float*)(smem + cfg::OFF_W3);

    const int tid  = threadIdx.x;
    const int lane = tid & 31;
    const int wave = tid >> 5;
    char* wb = smem + cfg::OFF_WV + wave * cfg::WVB;
    _Float16* actT = (_Float16*)(wb + cfg::WB_ACT);
    _Float16* c1T  = (_Float16*)(wb + cfg::WB_C1);
    float*    xf   = (float*)(wb + cfg::WB_XF);
    float*    red  = (float*)(wb + cfg::WB_RED);

    // ---- cooperative weight staging (block-wide) ----
    for (int i = tid; i < 5 * 128; i += 256) sW1[i] = W1[i];
    if (tid < 128) { sB1[tid] = b1[tid]; sB2[tid] = b2[tid]; sW3[tid] = W3[tid]; }
    for (int i = tid; i < 8 * 4 * 32 * 16; i += 256) {
        int j  = i & 15;
        int ln = (i >> 4) & 31;
        int ks = (i >> 9) & 3;
        int nt = i >> 11;
        int K  = ks * 32 + ((j >> 3) << 4) + ((ln >> 4) << 3) + (j & 7);   // B layout: lane half h: j<8 -> K=8h+j, j>=8 -> K=16+8h+(j-8)
        int n  = nt * 16 + (ln & 15);
        sW2f[i] = (_Float16)W2[K * 128 + n];       // forward: B = W2[k][n]
        sW2t[i] = (_Float16)W2[n * 128 + K];       // backward: B = W2^T[k][n] = W2[n][k]
    }
    __syncthreads();

    const int m16 = lane & 15;
    const int kh  = lane >> 4;
    const int p0  = blockIdx.x * cfg::PTS_B + wave * cfg::PTS_W;

    // ---- trilerp + feature staging (lanes 0..15, one point each) ----
    float gc[2][2][2][2];   // corner values, kept live for backward
    float fx = 0.f, fy = 0.f, fz = 0.f;
    if (lane < 16) {
        int p = p0 + m16;
        float px = pos[p * 3 + 0], py = pos[p * 3 + 1], pz = pos[p * 3 + 2];
        float ux = (px + 1.f) * 7.5f, uy = (py + 1.f) * 7.5f, uz = (pz + 1.f) * 7.5f;
        int ix = (int)floorf(ux); ix = ix < 0 ? 0 : (ix > 14 ? 14 : ix);
        int iy = (int)floorf(uy); iy = iy < 0 ? 0 : (iy > 14 ? 14 : iy);
        int iz = (int)floorf(uz); iz = iz < 0 ? 0 : (iz > 14 ? 14 : iz);
        fx = ux - (float)ix; fy = uy - (float)iy; fz = uz - (float)iz;
        float e0 = 0.f, e1 = 0.f;
#pragma unroll
        for (int dx = 0; dx < 2; ++dx)
#pragma unroll
            for (int dy = 0; dy < 2; ++dy)
#pragma unroll
                for (int dz = 0; dz < 2; ++dz) {
                    const float2 g = *(const float2*)(grid +
                        ((((ix + dx) * 16 + (iy + dy)) * 16 + (iz + dz)) * 2));
                    gc[dx][dy][dz][0] = g.x;
                    gc[dx][dy][dz][1] = g.y;
                    float w = (dx ? fx : 1.f - fx) * (dy ? fy : 1.f - fy) *
                              (dz ? fz : 1.f - fz);
                    e0 += w * g.x; e1 += w * g.y;
                }
        xf[m16 * 8 + 0] = e0; xf[m16 * 8 + 1] = e1;
        xf[m16 * 8 + 2] = px; xf[m16 * 8 + 3] = py; xf[m16 * 8 + 4] = pz;
        bool inb = px >= -1.f && px <= 1.f && py >= -1.f && py <= 1.f &&
                   pz >= -1.f && pz <= 1.f;
        out[4 * cfg::N + p] = inb ? 1.f : 0.f;     // mask
    }

    // ---- layer 1: z1 = x@W1+b1 ; h1=sin(30 z1); stash 30*cos(30 z1) ----
    for (int m = 0; m < 16; ++m) {
        float f0 = xf[m * 8 + 0], f1 = xf[m * 8 + 1], f2 = xf[m * 8 + 2];
        float f3 = xf[m * 8 + 3], f4 = xf[m * 8 + 4];
#pragma unroll
        for (int t = 0; t < 4; ++t) {
            int c = lane + 32 * t;
            float z = sB1[c] + f0 * sW1[c] + f1 * sW1[128 + c] + f2 * sW1[256 + c] +
                      f3 * sW1[384 + c] + f4 * sW1[512 + c];
            float arg = 30.f * z;
            actT[m * cfg::TS + c] = (_Float16)__sinf(arg);
            c1T[m * cfg::TS + c]  = (_Float16)(30.f * __cosf(arg));
        }
    }

    // ---- GEMM1: z2 = h1 @ W2 (WMMA), fused sin/W3-dot/gz2 store ----
    v16h a0 = load_a_frag(actT, m16, 0, kh);
    v16h a1 = load_a_frag(actT, m16, 32, kh);
    v16h a2 = load_a_frag(actT, m16, 64, kh);
    v16h a3 = load_a_frag(actT, m16, 96, kh);
    float sdfp[8];
#pragma unroll
    for (int r = 0; r < 8; ++r) sdfp[r] = 0.f;
#pragma unroll
    for (int nt = 0; nt < 8; ++nt) {
        v8f acc = {};
        acc = wmma16(a0, load_b_frag(sW2f, nt, 0, lane), acc);
        acc = wmma16(a1, load_b_frag(sW2f, nt, 1, lane), acc);
        acc = wmma16(a2, load_b_frag(sW2f, nt, 2, lane), acc);
        acc = wmma16(a3, load_b_frag(sW2f, nt, 3, lane), acc);
        int n = nt * 16 + m16;
        float bb = sB2[n], w3 = sW3[n];
#pragma unroll
        for (int r = 0; r < 8; ++r) {
            float z2 = acc[r] + bb;
            sdfp[r] += __sinf(z2) * w3;                 // layer3 partial
            // gz2 = dsdf/dz2 = W3 * cos(z2); overwrite act tile (h1 already in regs)
            actT[(r + kh * 8) * cfg::TS + n] = (_Float16)(w3 * __cosf(z2));
        }
    }
    // sdf reduction across the 16 N-lanes
#pragma unroll
    for (int r = 0; r < 8; ++r) red[lane * 8 + r] = sdfp[r];
    if (lane < 16) {
        float s = b3[0];
        int base = (m16 < 8) ? 0 : 16;
        int r = m16 & 7;
#pragma unroll
        for (int j = 0; j < 16; ++j) s += red[(base + j) * 8 + r];
        out[p0 + m16] = s;                              // sdf
    }

    // ---- GEMM2: g1 = gz2 @ W2^T (WMMA), fused * 30*cos(30 z1) ----
    v16h g0 = load_a_frag(actT, m16, 0, kh);
    v16h g1f = load_a_frag(actT, m16, 32, kh);
    v16h g2 = load_a_frag(actT, m16, 64, kh);
    v16h g3 = load_a_frag(actT, m16, 96, kh);
#pragma unroll
    for (int nt = 0; nt < 8; ++nt) {
        v8f acc = {};
        acc = wmma16(g0,  load_b_frag(sW2t, nt, 0, lane), acc);
        acc = wmma16(g1f, load_b_frag(sW2t, nt, 1, lane), acc);
        acc = wmma16(g2,  load_b_frag(sW2t, nt, 2, lane), acc);
        acc = wmma16(g3,  load_b_frag(sW2t, nt, 3, lane), acc);
        int n = nt * 16 + m16;
#pragma unroll
        for (int r = 0; r < 8; ++r) {
            int mr = r + kh * 8;
            float c1 = (float)c1T[mr * cfg::TS + n];
            actT[mr * cfg::TS + n] = (_Float16)(acc[r] * c1);  // gz1
        }
    }

    // ---- dx = gz1 @ W1^T ; split n-range across lane halves ----
    {
        float d0 = 0.f, d1 = 0.f, d2 = 0.f, d3 = 0.f, d4 = 0.f;
        int nb = kh * 64;
        for (int n = nb; n < nb + 64; ++n) {
            float g = (float)actT[m16 * cfg::TS + n];
            d0 += g * sW1[n];        d1 += g * sW1[128 + n];
            d2 += g * sW1[256 + n];  d3 += g * sW1[384 + n];
            d4 += g * sW1[512 + n];
        }
        red[lane * 8 + 0] = d0; red[lane * 8 + 1] = d1; red[lane * 8 + 2] = d2;
        red[lane * 8 + 3] = d3; red[lane * 8 + 4] = d4;
    }
    if (lane < 16) {
        float de0 = red[m16 * 8 + 0] + red[(m16 + 16) * 8 + 0];
        float de1 = red[m16 * 8 + 1] + red[(m16 + 16) * 8 + 1];
        float gpx = red[m16 * 8 + 2] + red[(m16 + 16) * 8 + 2];
        float gpy = red[m16 * 8 + 3] + red[(m16 + 16) * 8 + 3];
        float gpz = red[m16 * 8 + 4] + red[(m16 + 16) * 8 + 4];
        // trilerp backward: d emb / d pos, du/dp = 7.5
#pragma unroll
        for (int dx = 0; dx < 2; ++dx)
#pragma unroll
            for (int dy = 0; dy < 2; ++dy)
#pragma unroll
                for (int dz = 0; dz < 2; ++dz) {
                    float wx = dx ? fx : 1.f - fx, sx = dx ? 1.f : -1.f;
                    float wy = dy ? fy : 1.f - fy, sy = dy ? 1.f : -1.f;
                    float wz = dz ? fz : 1.f - fz, sz = dz ? 1.f : -1.f;
                    float g = de0 * gc[dx][dy][dz][0] + de1 * gc[dx][dy][dz][1];
                    gpx += 7.5f * sx * wy * wz * g;
                    gpy += 7.5f * sy * wx * wz * g;
                    gpz += 7.5f * sz * wx * wy * g;
                }
        int p = p0 + m16;
        out[cfg::N + p * 3 + 0] = gpx;
        out[cfg::N + p * 3 + 1] = gpy;
        out[cfg::N + p * 3 + 2] = gpz;
    }
}

extern "C" void kernel_launch(void* const* d_in, const int* in_sizes, int n_in,
                              void* d_out, int out_size, void* d_ws, size_t ws_size,
                              hipStream_t stream) {
    (void)in_sizes; (void)n_in; (void)out_size; (void)d_ws; (void)ws_size;
    const float* pos  = (const float*)d_in[0];
    const float* grid = (const float*)d_in[1];
    const float* W1   = (const float*)d_in[2];
    const float* b1   = (const float*)d_in[3];
    const float* W2   = (const float*)d_in[4];
    const float* b2   = (const float*)d_in[5];
    const float* W3   = (const float*)d_in[6];
    const float* b3   = (const float*)d_in[7];
    siren_sdf_fused<<<cfg::BLOCKS, 256, cfg::SMEM, stream>>>(
        pos, grid, W1, b1, W2, b2, W3, b3, (float*)d_out);
}
